// GraphSAGE_40089224741075
// MI455X (gfx1250) — matrix operand order repaired and checked by reading, not compiled, not run on hardware
//
#include <hip/hip_runtime.h>
#include <hip/hip_bf16.h>

typedef __attribute__((ext_vector_type(16))) _Float16 v16h;
typedef __attribute__((ext_vector_type(8)))  float    v8f;

#define HIDO 128   // hidden width (output cols of both SAGE layers)

// ---------------------------------------------------------------------------
// degree: deg[dst[e]] += 1  (fp32 L2 atomics, non-returning)
// ---------------------------------------------------------------------------
__global__ void degree_kernel(const int* __restrict__ dst, float* __restrict__ deg, int E) {
    int t = blockIdx.x * blockDim.x + threadIdx.x;
    if (t < E)
        __hip_atomic_fetch_add(deg + dst[t], 1.0f, __ATOMIC_RELAXED, __HIP_MEMORY_SCOPE_AGENT);
}

// deg_inv in place: deg[i] = 1/max(deg[i],1)
__global__ void deginv_kernel(float* __restrict__ deg, int N) {
    int i = blockIdx.x * blockDim.x + threadIdx.x;
    if (i < N) deg[i] = 1.0f / fmaxf(deg[i], 1.0f);
}

// ---------------------------------------------------------------------------
// Pack weight W (HIDO x K, row-major, fp32) into per-lane WMMA B fragments.
// B tile (32x16 f16) layout: lane L holds column n0 + (L&15); halves e=0..7
// hold K = k0 + 8h + e, e=8..15 hold K = k0 + 16 + 8h + (e-8), h = L>=16.  Storage: ((tn*numK+ks)*32+lane)*16 + e.
// ---------------------------------------------------------------------------
__global__ void pack_w_kernel(const float* __restrict__ W, _Float16* __restrict__ frag, int K) {
    int tid  = blockIdx.x * blockDim.x + threadIdx.x;
    int e    = tid & 15;
    int lane = (tid >> 4) & 31;
    int rest = tid >> 9;
    int numK = K >> 5;
    int ks   = rest % numK;
    int tn   = rest / numK;
    int n = tn * 16 + (lane & 15);
    int k = ks * 32 + ((lane < 16) ? 0 : 8) + ((e < 8) ? e : (8 + e));
    frag[tid] = (_Float16)W[(size_t)n * K + k];
}

// ---------------------------------------------------------------------------
// Edge aggregation: one wave32 per edge; each lane moves FPL contiguous
// floats.  Gather feat[src] (L2-resident), scatter-add into agg[dst].
// ---------------------------------------------------------------------------
template <int FPL>   // floats per lane: 2 (F=64) or 4 (F=128)
__global__ void edge_agg_kernel(const float* __restrict__ feat,
                                const int* __restrict__ src,
                                const int* __restrict__ dst,
                                float* __restrict__ agg, int E) {
    const int F    = FPL * 32;
    int wid  = (blockIdx.x * blockDim.x + threadIdx.x) >> 5;
    int lane = threadIdx.x & 31;
    if (wid >= E) return;
    int s = __builtin_amdgcn_readfirstlane(src[wid]);
    int d = __builtin_amdgcn_readfirstlane(dst[wid]);
    const float* fp = feat + (size_t)s * F + lane * FPL;
    float*       ap = agg  + (size_t)d * F + lane * FPL;
    float v[FPL];
    if constexpr (FPL == 2) {
        float2 t = *(const float2*)fp; v[0] = t.x; v[1] = t.y;
    } else {
        float4 t = *(const float4*)fp; v[0] = t.x; v[1] = t.y; v[2] = t.z; v[3] = t.w;
    }
#pragma unroll
    for (int i = 0; i < FPL; ++i)
        __hip_atomic_fetch_add(ap + i, v[i], __ATOMIC_RELAXED, __HIP_MEMORY_SCOPE_AGENT);
}

// ---------------------------------------------------------------------------
// A-fragment loader (from LDS tile) for v_wmma_f32_16x16x32_f16.
// A tile (16x32 f16): lane L<16 -> row L, halves 0..7 = K k0+0..7,
// halves 8..15 = K k0+16..23; lane L>=16 -> row L-16, K ranges +8 / +24.
// Four ds_load_b128 per call (row stride padded -> bank-conflict free).
// ---------------------------------------------------------------------------
__device__ inline v16h load_a_frag_lds(const float* rowBase, int k0, bool hi, float scale) {
    int off = hi ? 8 : 0;
    const float4* p0 = (const float4*)(rowBase + k0 + off);
    const float4* p1 = (const float4*)(rowBase + k0 + off + 16);
    float4 f0 = p0[0], f1 = p0[1], f2 = p1[0], f3 = p1[1];
    v16h a;
    a[0]  = (_Float16)(f0.x * scale); a[1]  = (_Float16)(f0.y * scale);
    a[2]  = (_Float16)(f0.z * scale); a[3]  = (_Float16)(f0.w * scale);
    a[4]  = (_Float16)(f1.x * scale); a[5]  = (_Float16)(f1.y * scale);
    a[6]  = (_Float16)(f1.z * scale); a[7]  = (_Float16)(f1.w * scale);
    a[8]  = (_Float16)(f2.x * scale); a[9]  = (_Float16)(f2.y * scale);
    a[10] = (_Float16)(f2.z * scale); a[11] = (_Float16)(f2.w * scale);
    a[12] = (_Float16)(f3.x * scale); a[13] = (_Float16)(f3.y * scale);
    a[14] = (_Float16)(f3.z * scale); a[15] = (_Float16)(f3.w * scale);
    return a;
}

// ---------------------------------------------------------------------------
// SAGE linear: Out[m][n] = relu( (Aagg[m]*deginv[m]) @ Wl^T + bias + Aroot[m] @ Wr^T )
// Block = 256 threads = 8 wave32s = one 16-row x 128-col output stripe.
// The 16xK agg/root tiles are staged global->LDS ONCE per block via
// global_load_async_to_lds_b128 (ASYNCcnt), then every wave builds its WMMA
// A-fragments from LDS.  Requires N % 16 == 0 (N=50000 -> grid 3125).
// ---------------------------------------------------------------------------
#define MAXROWP (HIDO + 4)                 // padded row stride (floats), K<=128
__global__ void sage_gemm_kernel(const float* __restrict__ Aagg,
                                 const float* __restrict__ deginv,
                                 const float* __restrict__ Aroot,
                                 const _Float16* __restrict__ fragL,
                                 const _Float16* __restrict__ fragR,
                                 const float* __restrict__ bias,
                                 float* __restrict__ Out,
                                 int K, int numK) {
    __shared__ float tile[2 * 16 * MAXROWP];    // [agg | root], padded rows
    const int ROWP = K + 4;
    const int lane = threadIdx.x & 31;
    const int tn   = threadIdx.x >> 5;          // 16-col tile (0..7)
    const int m0   = blockIdx.x * 16;
    const bool hi  = lane >= 16;

    // ---- stage A tiles global -> LDS (async, no VGPR round-trip) ----
    // chunk = 16 bytes; chunksPerMat = 16*K/4 (256 or 512) -> uniform loops.
    const int chunksPerMat = (16 * K) >> 2;
    for (int m = 0; m < 2; ++m) {
        const float* Abase = m ? Aroot : Aagg;
        float*       ldsB  = tile + m * 16 * ROWP;
        for (int ci = threadIdx.x; ci < chunksPerMat; ci += 256) {
            int row = ci / (K >> 2);
            int cb  = (ci % (K >> 2)) << 2;     // float index within row
            const float* g = Abase + (size_t)(m0 + row) * K + cb;
            unsigned lAddr = (unsigned)(size_t)(ldsB + row * ROWP + cb);
            asm volatile("global_load_async_to_lds_b128 %0, %1, off"
                         :: "v"(lAddr), "v"((const void*)g) : "memory");
        }
    }
    asm volatile("s_wait_asynccnt 0x0" ::: "memory");
    __syncthreads();

    const int rowL = lane & 15;
    const float sc = deginv[m0 + rowL];
    const float* aggRow  = tile + rowL * ROWP;
    const float* rootRow = tile + 16 * ROWP + rowL * ROWP;

    v8f acc = {};
    for (int ks = 0; ks < numK; ++ks) {
        int k0 = ks * 32;
        size_t fOff = ((size_t)(tn * numK + ks) * 32 + lane) << 4;
        v16h aA = load_a_frag_lds(aggRow, k0, hi, sc);
        v16h bL = *(const v16h*)(fragL + fOff);
        acc = __builtin_amdgcn_wmma_f32_16x16x32_f16(false, aA, false, bL,
                                                     (short)0, acc, false, false);
        v16h aR = load_a_frag_lds(rootRow, k0, hi, 1.0f);
        v16h bR = *(const v16h*)(fragR + fOff);
        acc = __builtin_amdgcn_wmma_f32_16x16x32_f16(false, aR, false, bR,
                                                     (short)0, acc, false, false);
    }

    // C/D layout: lanes 0-15 -> rows m0+v, lanes 16-31 -> rows m0+8+v;
    // col = tn*16 + lane&15.  Branchless coalesced epilogue.
    const int col = tn * 16 + rowL;
    const float b = bias[col];
    float* outp = Out + (size_t)(m0 + (hi ? 8 : 0)) * HIDO + col;
#pragma unroll
    for (int v = 0; v < 8; ++v)
        outp[(size_t)v * HIDO] = fmaxf(acc[v] + b, 0.0f);
}

// ---------------------------------------------------------------------------
// Final FC: out[n][0..1] = h1[n] @ Wfc^T + bfc   (OUT=2, trivial)
// ---------------------------------------------------------------------------
__global__ void fc_kernel(const float* __restrict__ h1, const float* __restrict__ Wfc,
                          const float* __restrict__ bfc, float* __restrict__ out, int N) {
    int n = blockIdx.x * blockDim.x + threadIdx.x;
    if (n >= N) return;
    const float4* hp = (const float4*)(h1 + (size_t)n * HIDO);
    const float4* w0 = (const float4*)(Wfc);
    const float4* w1 = (const float4*)(Wfc + HIDO);
    float a0 = bfc[0], a1 = bfc[1];
#pragma unroll 4
    for (int i = 0; i < HIDO / 4; ++i) {
        float4 h = hp[i], x0 = w0[i], x1 = w1[i];
        a0 += h.x * x0.x + h.y * x0.y + h.z * x0.z + h.w * x0.w;
        a1 += h.x * x1.x + h.y * x1.y + h.z * x1.z + h.w * x1.w;
    }
    out[n * 2 + 0] = a0;
    out[n * 2 + 1] = a1;
}

// ---------------------------------------------------------------------------
extern "C" void kernel_launch(void* const* d_in, const int* in_sizes, int n_in,
                              void* d_out, int out_size, void* d_ws, size_t ws_size,
                              hipStream_t stream) {
    const float* x   = (const float*)d_in[0];
    const int*   ei  = (const int*)d_in[1];
    const float* Wl0 = (const float*)d_in[2];
    const float* bl0 = (const float*)d_in[3];
    const float* Wr0 = (const float*)d_in[4];
    const float* Wl1 = (const float*)d_in[5];
    const float* bl1 = (const float*)d_in[6];
    const float* Wr1 = (const float*)d_in[7];
    const float* Wfc = (const float*)d_in[8];
    const float* bfc = (const float*)d_in[9];
    float* out = (float*)d_out;

    const int IN = 64;
    const int N  = in_sizes[0] / IN;       // 50000
    const int E  = in_sizes[1] / 2;        // 800000
    const int* src = ei;
    const int* dst = ei + E;

    // ---- workspace carve-up (256B aligned) ----
    char* ws = (char*)d_ws;
    size_t off = 0;
    auto carve = [&](size_t bytes) -> void* {
        void* p = ws + off;
        off = (off + bytes + 255) & ~(size_t)255;
        return p;
    };
    float*     deg  = (float*)carve((size_t)N * 4);            // becomes deg_inv
    float*     agg0 = (float*)carve((size_t)N * IN * 4);
    float*     h0   = (float*)carve((size_t)N * HIDO * 4);
    float*     agg1 = (float*)carve((size_t)N * HIDO * 4);
    float*     h1   = (float*)carve((size_t)N * HIDO * 4);
    _Float16*  fWl0 = (_Float16*)carve((size_t)HIDO * IN * 2);
    _Float16*  fWr0 = (_Float16*)carve((size_t)HIDO * IN * 2);
    _Float16*  fWl1 = (_Float16*)carve((size_t)HIDO * HIDO * 2);
    _Float16*  fWr1 = (_Float16*)carve((size_t)HIDO * HIDO * 2);

    // ---- zero the accumulation buffers (graph-capturable stream ops) ----
    hipMemsetAsync(deg,  0, (size_t)N * 4,        stream);
    hipMemsetAsync(agg0, 0, (size_t)N * IN * 4,   stream);
    hipMemsetAsync(agg1, 0, (size_t)N * HIDO * 4, stream);

    // ---- degree + inverse ----
    degree_kernel<<<(E + 255) / 256, 256, 0, stream>>>(dst, deg, E);
    deginv_kernel<<<(N + 255) / 256, 256, 0, stream>>>(deg, N);

    // ---- pack weights into WMMA B fragments (f32 -> f16) ----
    pack_w_kernel<<<(HIDO * IN)   / 256, 256, 0, stream>>>(Wl0, fWl0, IN);
    pack_w_kernel<<<(HIDO * IN)   / 256, 256, 0, stream>>>(Wr0, fWr0, IN);
    pack_w_kernel<<<(HIDO * HIDO) / 256, 256, 0, stream>>>(Wl1, fWl1, HIDO);
    pack_w_kernel<<<(HIDO * HIDO) / 256, 256, 0, stream>>>(Wr1, fWr1, HIDO);

    // ---- layer 0: aggregate x, then WMMA linear + ReLU ----
    edge_agg_kernel<2><<<(E + 7) / 8, 256, 0, stream>>>(x, src, dst, agg0, E);
    sage_gemm_kernel<<<N / 16, 256, 0, stream>>>(agg0, deg, x, fWl0, fWr0, bl0,
                                                 h0, IN, IN / 32);

    // ---- layer 1: aggregate h0, then WMMA linear + ReLU ----
    edge_agg_kernel<4><<<(E + 7) / 8, 256, 0, stream>>>(h0, src, dst, agg1, E);
    sage_gemm_kernel<<<N / 16, 256, 0, stream>>>(agg1, deg, h0, fWl1, fWr1, bl1,
                                                 h1, HIDO, HIDO / 32);

    // ---- final FC ----
    fc_kernel<<<(N + 255) / 256, 256, 0, stream>>>(h1, Wfc, bfc, out, N);
}
